// EfficientZeroBlock_63385127355007
// MI455X (gfx1250) — matrix operand order repaired and checked by reading, not compiled, not run on hardware
//
#include <hip/hip_runtime.h>

typedef __attribute__((ext_vector_type(16))) _Float16 v16h;
typedef __attribute__((ext_vector_type(8)))  _Float16 v8h;
typedef __attribute__((ext_vector_type(8)))  float    v8f;
typedef __attribute__((ext_vector_type(4)))  unsigned int u32x4;
typedef __attribute__((ext_vector_type(8)))  int      i32x8;
typedef __attribute__((ext_vector_type(4)))  int      i32x4;

#define B_    64
#define C_    64
#define T_    512
#define V_    25
#define K_    3
#define O_    64
#define TT    16            // t-tile per block
#define NJ    (TT*V_)       // 400 flattened (t,v) columns per tile
#define NTILE (NJ/16)       // 25 column tiles (exact)
#define CPAD  72            // padded channel stride in LDS (multiple of 8)

#if __has_builtin(__builtin_amdgcn_tensor_load_to_lds) && \
    __has_builtin(__builtin_amdgcn_s_wait_tensorcnt)
#define HAVE_TDM 1
#else
#define HAVE_TDM 0
#endif

__device__ __forceinline__ float hswish(float x){
  float c = fminf(fmaxf(x + 3.0f, 0.0f), 6.0f);
  return x * c * (1.0f/6.0f);
}

__device__ __forceinline__ v16h cat8(v8h lo, v8h hi){
  return __builtin_shufflevector(lo, hi, 0,1,2,3,4,5,6,7,8,9,10,11,12,13,14,15);
}

// A-operand fragment (16x32 f16). Source row-major rows[m][K], stride in halves.
// Lane L: m = L&15 ; element e -> K = (e&7) + 16*(e>>3) + 8*(L>>4)
__device__ __forceinline__ v16h load_a(const _Float16* base, int stride){
  int L = threadIdx.x & 31;
  const _Float16* p = base + (L & 15) * stride + ((L >> 4) << 3);
  v8h lo = *(const v8h*)(p);        // K = kh .. kh+7
  v8h hi = *(const v8h*)(p + 16);   // K = 16+kh .. 16+kh+7
  return cat8(lo, hi);
}
// B-operand fragment (32x16 f16). Source stored column-major cols[n][K], stride halves.
// Lane L: n = L&15 ; element e -> K = (e&7) + 16*(e>>3) + 8*(L>>4)
__device__ __forceinline__ v16h load_b(const _Float16* base, int stride){
  int L = threadIdx.x & 31;
  const _Float16* p = base + (L & 15) * stride + ((L >> 4) << 3);
  v8h lo = *(const v8h*)(p);
  v8h hi = *(const v8h*)(p + 16);
  return cat8(lo, hi);
}

#define WMMA_F16(a,b,c) __builtin_amdgcn_wmma_f32_16x16x32_f16(false,(a),false,(b),(short)0,(c),false,false)

// ---------------------------------------------------------------------------
// Kernel 1: adjacency normalization + weight conversion (tiny, one block)
// ---------------------------------------------------------------------------
__global__ __launch_bounds__(256) void prep_kernel(
    const float* __restrict__ Af, const float* __restrict__ Al,
    const float* __restrict__ Wg, const float* __restrict__ pw,
    _Float16* __restrict__ wsAkT, _Float16* __restrict__ wsWh,
    _Float16* __restrict__ wsPw)
{
  __shared__ float dsh[K_*V_];
  int tid = threadIdx.x;
  if (tid < K_*V_){
    int k = tid / V_, v = tid % V_;
    const float* fa = Af + (k*V_ + v)*V_;
    const float* la = Al + (k*V_ + v)*V_;
    float s = 0.0f;
    for (int w = 0; w < V_; ++w) s += fabsf(fa[w] + la[w]);
    s = fmaxf(s, 1e-6f);
    dsh[tid] = rsqrtf(s);
  }
  __syncthreads();
  // AkT[k][w][v] (transposed, zero-padded to 32x32), with 1/K folded in
  for (int i = tid; i < K_*32*32; i += 256){
    int k = i / 1024, r = i % 1024, w = r >> 5, v = r & 31;
    float val = 0.0f;
    if (w < V_ && v < V_){
      float a = fabsf(Af[(k*V_+v)*V_ + w] + Al[(k*V_+v)*V_ + w]);
      val = dsh[k*V_+v] * a * dsh[k*V_+w] * (1.0f/(float)K_);
    }
    wsAkT[i] = (_Float16)val;
  }
  for (int i = tid; i < K_*O_*C_; i += 256) wsWh[i] = (_Float16)Wg[i];
  for (int i = tid; i < O_*C_;    i += 256) wsPw[i] = (_Float16)pw[i];
}

// ---------------------------------------------------------------------------
// Kernel 2: fused GCN (channel-mix GEMM + V-mix GEMM over K partitions)
//           + BN + hardswish  -> yh (f16) in ws
// grid = (T/TT, B), 256 threads (8 waves)
// ---------------------------------------------------------------------------
__global__ __launch_bounds__(256) void gcn_kernel(
    const float* __restrict__ x, const _Float16* __restrict__ wsAkT,
    const _Float16* __restrict__ wsWh, const float* __restrict__ gsc,
    const float* __restrict__ gbi, _Float16* __restrict__ yh)
{
  extern __shared__ char smem[];
  _Float16* xs  = (_Float16*)smem;          // [400][72]   x tile, transposed (tv, c)
  _Float16* Z   = xs  + NJ*CPAD;            // [64][16][32] per-k channel-mixed tile
  _Float16* WhS = Z   + O_*TT*32;           // [3][64][64]
  _Float16* AkS = WhS + K_*O_*C_;           // [3][32][32]
  float*    bnS = (float*)(AkS + K_*32*32); // [128] scale|bias

  int tid  = threadIdx.x;
  int wave = tid >> 5;
  int L    = tid & 31;
  int t0   = blockIdx.x * TT;
  int b    = blockIdx.y;

  // stage x transposed: xs[j][c], j = (t-t0)*25+v  (t,v contiguous in memory)
  const float* xb = x + (size_t)b * C_ * T_ * V_ + (size_t)t0 * V_;
  for (int i = tid; i < C_*NJ; i += 256){
    int c = i / NJ, j = i % NJ;
    xs[j*CPAD + c] = (_Float16)xb[(size_t)c * (T_*V_) + j];
  }
  for (int i = tid; i < K_*O_*C_;  i += 256) WhS[i] = wsWh[i];
  for (int i = tid; i < K_*32*32;  i += 256) AkS[i] = wsAkT[i];
  for (int i = tid; i < O_*TT*32;  i += 256) Z[i]   = (_Float16)0.0f; // zero v-padding
  if (tid < 64)  bnS[tid]       = gsc[tid];
  if (tid < 64)  bnS[64 + tid]  = gbi[tid];
  __syncthreads();

  // V-mix accumulators: wave owns o = wave*8 .. wave*8+7, two w-tiles each
  v8f zero = {};
  v8f acc[8][2];
  #pragma unroll
  for (int i = 0; i < 8; ++i){ acc[i][0] = zero; acc[i][1] = zero; }

  for (int k = 0; k < K_; ++k){
    // --- GEMM1: Z[o][t][v] = sum_c Wk[o][c] * x[(t,v)][c] ---
    for (int nt = wave; nt < NTILE; nt += 8){
      v16h bf0 = load_b(xs + nt*16*CPAD, CPAD);        // c 0..31
      v16h bf1 = load_b(xs + nt*16*CPAD + 32, CPAD);   // c 32..63
      int j = nt*16 + (L & 15);
      int t = j / V_, v = j % V_;
      int mhalf = (L >> 4) << 3;
      #pragma unroll
      for (int mt = 0; mt < 4; ++mt){
        const _Float16* wkb = WhS + k*O_*C_ + mt*16*C_;
        v16h a0 = load_a(wkb,      C_);
        v16h a1 = load_a(wkb + 32, C_);
        v8f d = zero;
        d = WMMA_F16(a0, bf0, d);
        d = WMMA_F16(a1, bf1, d);
        #pragma unroll
        for (int r = 0; r < 8; ++r){
          int o = mt*16 + mhalf + r;           // D: M = r + 8*(L>>4)
          Z[(o*TT + t)*32 + v] = (_Float16)d[r];
        }
      }
    }
    __syncthreads();
    // --- GEMM2 (V-mix): acc[(o,t)][w] += Z[o][t][v] * AkT[w][v] ---
    v16h bw0 = load_b(AkS + k*1024,          32);      // w 0..15
    v16h bw1 = load_b(AkS + k*1024 + 16*32,  32);      // w 16..31 (padded)
    #pragma unroll
    for (int i = 0; i < 8; ++i){
      int o = wave*8 + i;
      v16h a = load_a(Z + o*TT*32, 32);                // rows = t, K = v (padded 32)
      acc[i][0] = WMMA_F16(a, bw0, acc[i][0]);
      acc[i][1] = WMMA_F16(a, bw1, acc[i][1]);
    }
    __syncthreads();   // protect Z before next k overwrites it
  }

  // epilogue: BN + hardswish, store f16 y[b][o][t][w]
  #pragma unroll
  for (int i = 0; i < 8; ++i){
    int o = wave*8 + i;
    float sc = bnS[o], bi = bnS[64 + o];
    #pragma unroll
    for (int wt = 0; wt < 2; ++wt){
      int w = wt*16 + (L & 15);
      if (w < V_){
        int thalf = (L >> 4) << 3;
        #pragma unroll
        for (int r = 0; r < 8; ++r){
          int t = thalf + r;
          float val = hswish(acc[i][wt][r] * sc + bi);
          yh[(((size_t)b*O_ + o)*T_ + (t0 + t))*V_ + w] = (_Float16)val;
        }
      }
    }
  }
}

// ---------------------------------------------------------------------------
// Kernel 3: depthwise temporal conv(k=9,pad4) + BN + hswish  (VALU)
//           + pointwise 64x64 GEMM (WMMA) + BN + residual + hswish -> out
// Interior tiles stage the y-halo tile with one TDM tensor_load_to_lds
// (D# per CDNA5 ISA: 2-D tile 600x64 halves, row stride T*V).
// grid = (T/TT, B), 256 threads
// ---------------------------------------------------------------------------
__global__ __launch_bounds__(256) void tcn_kernel(
    const _Float16* __restrict__ yh, const float* __restrict__ x,
    const float* __restrict__ dw, const _Float16* __restrict__ wsPw,
    const float* __restrict__ s1, const float* __restrict__ b1,
    const float* __restrict__ s2, const float* __restrict__ b2,
    float* __restrict__ out)
{
  extern __shared__ char smem[];
  _Float16* ys  = (_Float16*)smem;          // [64][24][25]  y with +-4 t halo
  _Float16* sA  = ys  + O_*24*V_;           // [400][72]     conv output, transposed
  _Float16* pwS = sA  + NJ*CPAD;            // [64][64]
  float*    dwS = (float*)(pwS + O_*C_);    // [64][9]
  float*    bnS = dwS + O_*9;               // [256] s1|b1|s2|b2

  int tid  = threadIdx.x;
  int wave = tid >> 5;
  int L    = tid & 31;
  int t0   = blockIdx.x * TT;
  int b    = blockIdx.y;

  bool interior = (blockIdx.x > 0) && (blockIdx.x + 1 < gridDim.x);

#if HAVE_TDM
  if (interior){
    if (tid < 32){   // single wave issues the tensor DMA (EXEC-independent op)
      // source: yh[b][o][t0-4 .. t0+20)[0..25): 64 rows x 600 halves, row
      // stride T*V halves -> lands contiguously in ys[64][600]
      unsigned long long gaddr =
          (unsigned long long)(const void*)(yh + ((size_t)b*O_*T_ + (size_t)(t0 - 4))*V_);
      unsigned int laddr = (unsigned int)(unsigned long long)(void*)ys;  // LDS byte offset
      u32x4 g0 = {0u, 0u, 0u, 0u};
      g0[0] = 1u;                                        // count=1, user descriptor
      g0[1] = laddr;                                     // lds_addr
      g0[2] = (unsigned int)(gaddr & 0xffffffffull);     // global_addr[31:0]
      g0[3] = (unsigned int)((gaddr >> 32) & 0x1ffffffull) | (2u << 30); // addr[56:32] | type=2
      i32x8 g1 = {0,0,0,0,0,0,0,0};
      g1[0] = (1 << 16);                                 // data_size = 2 bytes
      g1[1] = (600 << 16);                               // tensor_dim0[15:0] = 600
      g1[2] = (64 << 16);                                // tensor_dim1[15:0] = 64
      g1[3] = (600 << 16);                               // tile_dim0 = 600
      g1[4] = 64;                                        // tile_dim1 = 64, tile_dim2 = 0
      g1[5] = (int)(T_ * V_);                            // tensor_dim0_stride = 12800
      i32x4 g2 = {0,0,0,0};
      i32x4 g3 = {0,0,0,0};
#if defined(__clang_major__) && (__clang_major__ >= 23)
      i32x8 gx = {0,0,0,0,0,0,0,0};
      __builtin_amdgcn_tensor_load_to_lds(g0, g1, g2, g3, gx, 0);
#else
      __builtin_amdgcn_tensor_load_to_lds(g0, g1, g2, g3, 0);
#endif
      __builtin_amdgcn_s_wait_tensorcnt(0);
    }
  } else
#endif
  {
    for (int i = tid; i < O_*24*V_; i += 256){
      int o = i / (24*V_), r = i % (24*V_), tl = r / V_, v = r % V_;
      int t = t0 - 4 + tl;
      _Float16 val = (_Float16)0.0f;
      if (t >= 0 && t < T_) val = yh[(((size_t)b*O_ + o)*T_ + t)*V_ + v];
      ys[i] = val;
    }
  }
  for (int i = tid; i < O_*C_; i += 256) pwS[i] = wsPw[i];
  for (int i = tid; i < O_*9;  i += 256) dwS[i] = dw[i];
  if (tid < 64){
    bnS[tid]        = s1[tid];
    bnS[64  + tid]  = b1[tid];
    bnS[128 + tid]  = s2[tid];
    bnS[192 + tid]  = b2[tid];
  }
  __syncthreads();

  // depthwise temporal conv + BN1 + hardswish -> sA[(t,v)][o]  (f16, transposed)
  for (int i = tid; i < O_*NJ; i += 256){
    int o = i / NJ, j = i % NJ;
    int t = j / V_, v = j % V_;
    const _Float16* yp = ys + (o*24 + t)*V_ + v;   // taps at tl = t..t+8
    const float*    dp = dwS + o*9;
    float a = 0.0f;
    #pragma unroll
    for (int u = 0; u < 9; ++u) a += (float)yp[u*V_] * dp[u];
    a = hswish(a * bnS[o] + bnS[64 + o]);
    sA[j*CPAD + o] = (_Float16)a;
  }
  __syncthreads();

  // pointwise GEMM: out[o][(t,v)] = sum_c pw[o][c] * sA[(t,v)][c], + BN2 + res + hswish
  v8f zero = {};
  for (int nt = wave; nt < NTILE; nt += 8){
    v16h bf0 = load_b(sA + nt*16*CPAD, CPAD);        // c 0..31
    v16h bf1 = load_b(sA + nt*16*CPAD + 32, CPAD);   // c 32..63
    int j = nt*16 + (L & 15);
    int t = j / V_, v = j % V_;
    int mhalf = (L >> 4) << 3;
    #pragma unroll
    for (int mt = 0; mt < 4; ++mt){
      v16h a0 = load_a(pwS + mt*16*C_,      C_);
      v16h a1 = load_a(pwS + mt*16*C_ + 32, C_);
      v8f d = zero;
      d = WMMA_F16(a0, bf0, d);
      d = WMMA_F16(a1, bf1, d);
      #pragma unroll
      for (int r = 0; r < 8; ++r){
        int o = mt*16 + mhalf + r;
        size_t idx = (((size_t)b*O_ + o)*T_ + (t0 + t))*V_ + v;
        float val = d[r] * bnS[128 + o] + bnS[192 + o] + x[idx];
        out[idx] = hswish(val);
      }
    }
  }
}

// ---------------------------------------------------------------------------
extern "C" void kernel_launch(void* const* d_in, const int* in_sizes, int n_in,
                              void* d_out, int out_size, void* d_ws, size_t ws_size,
                              hipStream_t stream)
{
  (void)in_sizes; (void)n_in; (void)out_size; (void)ws_size;
  const float* x   = (const float*)d_in[0];
  const float* Af  = (const float*)d_in[1];
  const float* Al  = (const float*)d_in[2];
  const float* Wg  = (const float*)d_in[3];
  const float* gsc = (const float*)d_in[4];
  const float* gbi = (const float*)d_in[5];
  const float* dw  = (const float*)d_in[6];
  const float* pw  = (const float*)d_in[7];
  const float* s1  = (const float*)d_in[8];
  const float* b1  = (const float*)d_in[9];
  const float* s2  = (const float*)d_in[10];
  const float* b2  = (const float*)d_in[11];
  float* out = (float*)d_out;

  char* ws = (char*)d_ws;
  _Float16* wsAkT = (_Float16*)ws;                     // 3*32*32 halves
  _Float16* wsWh  = wsAkT + K_*32*32;                  // 3*64*64
  _Float16* wsPw  = wsWh  + K_*O_*C_;                  // 64*64
  _Float16* yh    = (_Float16*)(ws + 65536);           // B*O*T*V halves (~100 MB)

  prep_kernel<<<dim3(1), dim3(256), 0, stream>>>(Af, Al, Wg, pw, wsAkT, wsWh, wsPw);

  size_t sm2 = (size_t)(NJ*CPAD + O_*TT*32 + K_*O_*C_ + K_*32*32) * sizeof(_Float16)
             + 128 * sizeof(float);
  gcn_kernel<<<dim3(T_/TT, B_), dim3(256), sm2, stream>>>(x, wsAkT, wsWh, gsc, gbi, yh);

  size_t sm3 = (size_t)(O_*24*V_ + NJ*CPAD + O_*C_) * sizeof(_Float16)
             + (O_*9 + 256) * sizeof(float);
  tcn_kernel<<<dim3(T_/TT, B_), dim3(256), sm3, stream>>>(yh, x, dw, wsPw,
                                                          s1, b1, s2, b2, out);
}
